// LSTM_49349174231677
// MI455X (gfx1250) — compile-verified, matrix operand order checked by reading, not run on hardware
//
#include <hip/hip_runtime.h>

typedef __attribute__((ext_vector_type(16))) __bf16 v16bf;
typedef __attribute__((ext_vector_type(8)))  __bf16 v8bf;
typedef __attribute__((ext_vector_type(8)))  float  v8f;
typedef __attribute__((ext_vector_type(4)))  float  v4f;

#define T_STEPS 512
#define BATCH   64
#define HID     1024
#define KDIM    2048          // F + H
#define GATES   4096          // 4*H
#define NBLOCKS 16

// ---------------------------------------------------------------------------
// Fast activations: v_exp_f32 + v_rcp_f32, no IEEE div expansion
// ---------------------------------------------------------------------------
__device__ __forceinline__ float fast_sigmoid(float x) {
    float e = __expf(-x);
    return __builtin_amdgcn_rcpf(1.0f + e);
}
__device__ __forceinline__ float fast_tanh(float x) {
    float e = __expf(-2.0f * fabsf(x));
    float r = 1.0f - 2.0f * e * __builtin_amdgcn_rcpf(1.0f + e);   // (1-e)/(1+e)
    return copysignf(r, x);
}

// ---------------------------------------------------------------------------
// Kernel 0: repack [W_ih | W_hh] (fp32, row-major [4096 x 1024] each) into the
// bf16 WMMA B-operand chunk layout:
//   chunk linear id c = ((group*64 + kk)*2 + ch)*32 + lane, 8 bf16 per chunk
//   group = gate_row/16, kk = k/32, ch = (k%32)/16, lane encodes row%16, k%16/8
// A wave's B-tile load is then 32 consecutive 16B chunks -> fully coalesced.
// ---------------------------------------------------------------------------
__global__ void swizzle_weights(const float* __restrict__ W_ih,
                                const float* __restrict__ W_hh,
                                __bf16* __restrict__ W2s) {
    unsigned id = blockIdx.x * 256u + threadIdx.x;      // 0 .. 4096*2048-1
    unsigned elem  = id & 7u;
    unsigned lane  = (id >> 3) & 31u;
    unsigned ch    = (id >> 8) & 1u;
    unsigned kk    = (id >> 9) & 63u;
    unsigned group = id >> 15;                           // 0..255
    unsigned g = (group << 4) + (lane & 15u);            // gate row 0..4095
    unsigned k = (kk << 5) + (ch << 4) + ((lane >> 4) << 3) + elem; // 0..2047
    float v = (k < 1024u) ? W_ih[(size_t)g * 1024u + k]
                          : W_hh[(size_t)g * 1024u + (k - 1024u)];
    W2s[id] = (__bf16)v;
}

// ---------------------------------------------------------------------------
// Kernel 1: zero the h ping-pong buffer (h0 = 0) and barrier counters
// ---------------------------------------------------------------------------
__global__ void init_state(__bf16* __restrict__ hbuf, unsigned* __restrict__ bar) {
    unsigned id = blockIdx.x * 256u + threadIdx.x;       // 512 blocks -> 131072
    if (id < 2u * BATCH * HID) hbuf[id] = (__bf16)0.0f;
    if (id < T_STEPS) bar[id] = 0u;
}

// ---------------------------------------------------------------------------
// Kernel 2: persistent fused LSTM.
// grid = 16 blocks x 256 threads (8 wave32). Wave (cg, mh):
//   cg = blockIdx*4 + wave/2  -> hidden columns [cg*16, cg*16+16)
//   mh = wave&1               -> batch rows [mh*32, mh*32+32) (2 M-tiles)
// Per step: block stages X2 = [x_t | h_prev] as bf16 in LDS (A layout),
// each wave runs 4 gates x 2 Mtiles x 64 K-chunks of v_wmma_f32_16x16x32_bf16,
// then the LSTM cell elementwise. c stays in VGPRs across all 512 steps.
// ---------------------------------------------------------------------------
__global__ __launch_bounds__(256, 1)
void lstm_persistent(const float* __restrict__ seq,
                     const float* __restrict__ b_ih,
                     const float* __restrict__ b_hh,
                     const __bf16* __restrict__ W2s,
                     __bf16* __restrict__ hbuf,
                     unsigned* __restrict__ bar,
                     float* __restrict__ out) {
    extern __shared__ __bf16 lds[];   // 4 Mgroups * 64 kk * 2 ch * 32 lanes * 8 = 256 KB

    const int tid  = threadIdx.x;
    const int lane = tid & 31;
    const int wave = tid >> 5;
    const int cg   = blockIdx.x * 4 + (wave >> 1);   // 0..63
    const int mh   = wave & 1;                       // batch half
    const int col  = cg * 16 + (lane & 15);          // hidden column this lane owns
    const int mrow_hi = (lane >> 4) & 1;

    // per-lane bias for the 4 gate regions (depends on column only)
    float bias[4];
#pragma unroll
    for (int r = 0; r < 4; ++r) {
        int g = r * 1024 + col;
        bias[r] = b_ih[g] + b_hh[g];
    }

    // cell state for the 2 owned 16x16 patches, resident in VGPRs for all T
    v8f cst[2];
#pragma unroll
    for (int mt = 0; mt < 2; ++mt)
#pragma unroll
        for (int e = 0; e < 8; ++e) cst[mt][e] = 0.0f;

    for (int t = 0; t < T_STEPS; ++t) {
        const float*  xt    = seq  + (size_t)t * BATCH * 1024;
        const __bf16* hcur  = hbuf + (size_t)(t & 1)       * BATCH * HID;
        __bf16*       hnext = hbuf + (size_t)((t + 1) & 1) * BATCH * HID;

        // ---- stage X2 = [x_t | h_prev] into LDS in WMMA A-operand layout ----
        for (int c = tid; c < 16384; c += 256) {
            int lane_c = c & 31;
            int ch     = (c >> 5) & 1;
            int kk     = (c >> 6) & 63;
            int mg     = (c >> 12) & 3;
            int row    = mg * 16 + (lane_c & 15);
            int k      = kk * 32 + ch * 16 + ((lane_c >> 4) & 1) * 8;
            v8bf val;
            if (k < 1024) {                       // x part: fp32 -> bf16
                const float* s = xt + (size_t)row * 1024 + k;
                v4f a = *(const v4f*)s;
                v4f b = *(const v4f*)(s + 4);
                val[0] = (__bf16)a.x; val[1] = (__bf16)a.y;
                val[2] = (__bf16)a.z; val[3] = (__bf16)a.w;
                val[4] = (__bf16)b.x; val[5] = (__bf16)b.y;
                val[6] = (__bf16)b.z; val[7] = (__bf16)b.w;
            } else {                              // h part: already bf16
                val = *(const v8bf*)(hcur + (size_t)row * 1024 + (k - 1024));
            }
            *(v8bf*)(lds + (size_t)c * 8) = val;
        }
        __syncthreads();

        // ---- GEMM: 8 tiles (4 gates x 2 Mtiles), K = 2048 ----
        v8f acc[4][2];
#pragma unroll
        for (int r = 0; r < 4; ++r)
#pragma unroll
            for (int mt = 0; mt < 2; ++mt)
#pragma unroll
                for (int e = 0; e < 8; ++e) acc[r][mt][e] = bias[r];

        const int mgBase = mh * 2;
#pragma unroll 2
        for (int kk = 0; kk < 64; ++kk) {
            v16bf A[2];
#pragma unroll
            for (int mt = 0; mt < 2; ++mt) {
                const __bf16* p =
                    lds + ((size_t)(((mgBase + mt) * 64 + kk) * 2) * 32 + lane) * 8;
                v8bf lo = *(const v8bf*)p;
                v8bf hi = *(const v8bf*)(p + 256);   // ch=1 chunk
                A[mt] = __builtin_shufflevector(lo, hi, 0, 1, 2, 3, 4, 5, 6, 7,
                                                8, 9, 10, 11, 12, 13, 14, 15);
            }
#pragma unroll
            for (int r = 0; r < 4; ++r) {
                const int group = r * 64 + cg;
                const __bf16* p =
                    W2s + ((size_t)((group * 64 + kk) * 2) * 32 + lane) * 8;
                v8bf lo = *(const v8bf*)p;
                v8bf hi = *(const v8bf*)(p + 256);
                v16bf Bv = __builtin_shufflevector(lo, hi, 0, 1, 2, 3, 4, 5, 6, 7,
                                                   8, 9, 10, 11, 12, 13, 14, 15);
#pragma unroll
                for (int mt = 0; mt < 2; ++mt) {
                    acc[r][mt] = __builtin_amdgcn_wmma_f32_16x16x32_bf16(
                        false, A[mt], false, Bv, (short)0, acc[r][mt], false, false);
                }
            }
        }

        // ---- LSTM cell elementwise + output/state stores ----
        const size_t outHbase = (size_t)t * BATCH * HID;
        const size_t outCbase = (size_t)T_STEPS * BATCH * HID + outHbase;
#pragma unroll
        for (int mt = 0; mt < 2; ++mt) {
#pragma unroll
            for (int e = 0; e < 8; ++e) {
                float iv = fast_sigmoid(acc[0][mt][e]);
                float fv = fast_sigmoid(acc[1][mt][e]);
                float gv = fast_tanh(acc[2][mt][e]);
                float ov = fast_sigmoid(acc[3][mt][e]);
                float cv = fv * cst[mt][e] + iv * gv;
                float hv = ov * fast_tanh(cv);
                cst[mt][e] = cv;
                int b = (mh * 2 + mt) * 16 + e + mrow_hi * 8;   // D layout: M = e (+8)
                size_t off = (size_t)b * HID + col;
                __builtin_nontemporal_store(hv, out + outHbase + off);
                __builtin_nontemporal_store(cv, out + outCbase + off);
                hnext[off] = (__bf16)hv;
            }
        }

        // ---- device-scope step barrier (h broadcast before next staging) ----
        __threadfence();
        __syncthreads();
        if (tid == 0) {
            atomicAdd(&bar[t], 1u);
            volatile unsigned* vb = bar + t;
            while (*vb < (unsigned)NBLOCKS) { __builtin_amdgcn_s_sleep(1); }
        }
        __syncthreads();
        __threadfence();
    }
}

// ---------------------------------------------------------------------------
// Host-side launch
// ---------------------------------------------------------------------------
extern "C" void kernel_launch(void* const* d_in, const int* in_sizes, int n_in,
                              void* d_out, int out_size, void* d_ws, size_t ws_size,
                              hipStream_t stream) {
    const float* seq  = (const float*)d_in[0];
    const float* W_ih = (const float*)d_in[1];
    const float* W_hh = (const float*)d_in[2];
    const float* b_ih = (const float*)d_in[3];
    const float* b_hh = (const float*)d_in[4];

    char* ws = (char*)d_ws;
    __bf16*   W2s  = (__bf16*)ws;                                   // 16 MB
    __bf16*   hbuf = (__bf16*)(ws + (16u << 20));                   // 256 KB
    unsigned* bar  = (unsigned*)(ws + (16u << 20) + (256u << 10));  // 2 KB

    // one-time weight repack: 4096*2048 bf16 elements
    hipLaunchKernelGGL(swizzle_weights, dim3(32768), dim3(256), 0, stream,
                       W_ih, W_hh, W2s);
    // zero h0 ping-pong buffer + per-step barrier counters
    hipLaunchKernelGGL(init_state, dim3(512), dim3(256), 0, stream, hbuf, bar);
    // persistent fused LSTM: 16 blocks x 256 threads, 256 KB dynamic LDS each
    hipLaunchKernelGGL(lstm_persistent, dim3(NBLOCKS), dim3(256), 262144, stream,
                       seq, b_ih, b_hh, W2s, hbuf, bar, (float*)d_out);
}
